// MultiHeadAttention_87832081203977
// MI455X (gfx1250) — matrix operand order repaired and checked by reading, not compiled, not run on hardware
//
#include <hip/hip_runtime.h>
#include <hip/hip_bf16.h>

typedef __attribute__((ext_vector_type(16))) __bf16 v16bf;
typedef __attribute__((ext_vector_type(8)))  float  v8f;

union Frag16 {
    v16bf v;
    uint4 q[2];
    unsigned int w[8];
    unsigned short u[16];
};

// cheap fp32 -> bf16 (round-half-up; 1-ulp vs RNE, irrelevant next to bf16 quantization)
__device__ __forceinline__ unsigned short f2bf(float f) {
    unsigned u = __builtin_bit_cast(unsigned, f) + 0x8000u;
    return (unsigned short)(u >> 16);
}

// packed pair conversion: lo in bits [15:0], hi in bits [31:16].
// v_perm_b32 picks bytes {hi.b3, hi.b2, lo.b3, lo.b2} in ONE instruction.
__device__ __forceinline__ unsigned int f2bf2(float lo, float hi) {
    unsigned ulo = __builtin_bit_cast(unsigned, lo) + 0x8000u;
    unsigned uhi = __builtin_bit_cast(unsigned, hi) + 0x8000u;
#if __has_builtin(__builtin_amdgcn_perm)
    return __builtin_amdgcn_perm(uhi, ulo, 0x07060302u);
#else
    return (ulo >> 16) | (uhi & 0xFFFF0000u);
#endif
}

// ---------------------------------------------------------------------------
// C[M,N] = alpha * (A[M,K] @ B[K,N] + bias[N])   (A,B,C fp32; math bf16 WMMA)
// Block tile 64x256, 256 threads = 8 waves, each wave 32x64 (2x4 WMMA tiles),
// K step 32 -> 8 WMMA per wave per K-step; every A frag feeds 4 WMMAs.
// LDS: A row-major [64][32] bf16, B col-major [n][k] bf16 (contiguous frags).
// ---------------------------------------------------------------------------
#define TM 64
#define TN 256
#define TK 32

__global__ __launch_bounds__(256) void gemm_bias_bf16(
    const float* __restrict__ A, const float* __restrict__ Bw,
    const float* __restrict__ bias, float* __restrict__ C,
    int M, int N, int K, float alpha)
{
    __shared__ __align__(16) unsigned short Als[TM * TK];   // [row][k]
    __shared__ __align__(16) unsigned short Bls[TN * TK];   // [col][k]

    const int tid  = threadIdx.x;
    const int lane = tid & 31;
    const int wave = tid >> 5;          // 0..7
    const int wr   = wave >> 2;         // 0..1  (32-row strip)
    const int wc   = wave & 3;          // 0..3  (64-col strip)

    const int m0 = blockIdx.y * TM;
    const int n0 = blockIdx.x * TN;

    v8f acc[2][4];
#pragma unroll
    for (int i = 0; i < 2; ++i)
#pragma unroll
        for (int j = 0; j < 4; ++j)
            acc[i][j] = v8f{0.f, 0.f, 0.f, 0.f, 0.f, 0.f, 0.f, 0.f};

    // cooperative-load mapping
    const int ar  = tid >> 2;           // 0..63   A row
    const int ac  = (tid & 3) * 8;      // 0/8/16/24
    const int bk2 = (tid >> 4) * 2;     // 0..30   B k-pair
    const int bn0 = (tid & 15) * 16;    // 0..240  B col group

    const int nlan  = lane & 15;
    const int half8 = (lane >> 4) * 8;  // 0 or 8

    unsigned int* Als32 = (unsigned int*)Als;
    unsigned int* Bls32 = (unsigned int*)Bls;

    for (int kt = 0; kt < K; kt += TK) {
        // ---- stage A tile (fp32 -> bf16, packed-pair stores) ----
        const float* ag = A + (size_t)(m0 + ar) * K + kt + ac;
        float4 a0 = *(const float4*)(ag);
        float4 a1 = *(const float4*)(ag + 4);
        float av[8];
#pragma unroll
        for (int j = 0; j < 4; ++j) {
            av[j]     = ((const float*)&a0)[j];
            av[4 + j] = ((const float*)&a1)[j];
        }
#pragma unroll
        for (int j = 0; j < 4; ++j)
            Als32[(ar * TK + ac) / 2 + j] = f2bf2(av[2 * j], av[2 * j + 1]);

        // ---- stage B tile: thread owns 2 adjacent k-rows x 16 cols ----
        const float* bg = Bw + (size_t)(kt + bk2) * N + n0 + bn0;
        float4 r0[4], r1[4];
#pragma unroll
        for (int j2 = 0; j2 < 4; ++j2) {
            r0[j2] = *(const float4*)(bg + j2 * 4);
            r1[j2] = *(const float4*)(bg + N + j2 * 4);
        }
#pragma unroll
        for (int j = 0; j < 16; ++j)
            Bls32[((bn0 + j) * TK + bk2) / 2] =
                f2bf2(((const float*)r0)[j], ((const float*)r1)[j]);

        if (kt + TK < K) {              // hint next tiles into cache
            __builtin_prefetch(ag + TK, 0, 0);
            __builtin_prefetch(bg + (size_t)TK * N, 0, 0);
        }
        __syncthreads();

        // ---- fragments ----
        v16bf afr[2], bfr[4];
#pragma unroll
        for (int mi = 0; mi < 2; ++mi) {
            int row = wr * 32 + mi * 16 + nlan;
            Frag16 f;
            f.q[0] = *(const uint4*)&Als[row * TK + half8];        // K = ka..ka+7
            f.q[1] = *(const uint4*)&Als[row * TK + half8 + 16];   // K = ka+16..ka+23
            afr[mi] = f.v;
        }
#pragma unroll
        for (int ni = 0; ni < 4; ++ni) {
            int col = wc * 64 + ni * 16 + nlan;
            int kb  = half8;                                        // 0 or 8
            Frag16 f;
            f.q[0] = *(const uint4*)&Bls[col * TK + kb];
            f.q[1] = *(const uint4*)&Bls[col * TK + kb + 16];
            bfr[ni] = f.v;
        }
#pragma unroll
        for (int mi = 0; mi < 2; ++mi)
#pragma unroll
            for (int ni = 0; ni < 4; ++ni)
                acc[mi][ni] = __builtin_amdgcn_wmma_f32_16x16x32_bf16(
                    false, afr[mi], false, bfr[ni], (short)0, acc[mi][ni], false, false);

        __syncthreads();
    }

    // ---- epilogue: bias + alpha ----
#pragma unroll
    for (int ni = 0; ni < 4; ++ni) {
        int col = n0 + wc * 64 + ni * 16 + nlan;
        float bv = bias[col];
#pragma unroll
        for (int mi = 0; mi < 2; ++mi) {
#pragma unroll
            for (int i = 0; i < 8; ++i) {
                int row = m0 + wr * 32 + mi * 16 + half8 + i;
                C[(size_t)row * N + col] = alpha * (acc[mi][ni][i] + bv);
            }
        }
    }
}

// ---------------------------------------------------------------------------
// Flash attention over bf16 WMMA. Grid (Sq/64, nh, B); 128 threads = 4 waves,
// each wave owns 16 query rows. K tile: LDS row-major [k'][d]; V tile: LDS
// col-major [d][k'] so both B-fragment reads are contiguous ds_load_b128.
// Staging threads own 2 adjacent k' rows -> all LDS stores are packed 32-bit.
// Q/K/V/O all stored as [B, S, nh*dh] fp32 (GEMM-natural layout).
// ---------------------------------------------------------------------------
__global__ __launch_bounds__(128) void attn_kernel(
    const float* __restrict__ Qb, const float* __restrict__ Kb,
    const float* __restrict__ Vb, float* __restrict__ Ob)
{
    __shared__ __align__(16) unsigned short Kls[64 * 64];
    __shared__ __align__(16) unsigned short Vls[64 * 64];
    __shared__ __align__(16) unsigned short Pls[4 * 16 * 64];   // per-wave P slab

    const int tid  = threadIdx.x;
    const int lane = tid & 31;
    const int w    = tid >> 5;                      // wave 0..3
    const int h    = blockIdx.y;
    const int b    = blockIdx.z;
    const int q0   = blockIdx.x * 64 + w * 16;

    const size_t base_bh = ((size_t)b * 2048) * 1024 + (size_t)h * 64;

    const int nlan  = lane & 15;
    const int half8 = (lane >> 4) * 8;              // 0 or 8

    // ---- Q A-fragments (1/sqrt(64) folded into Q projection) ----
    v16bf aq[2];
    {
        const float* qp = Qb + base_bh + (size_t)(q0 + nlan) * 1024;
#pragma unroll
        for (int s = 0; s < 2; ++s) {
            int d0 = s * 32 + half8;
            float4 c0 = *(const float4*)(qp + d0);
            float4 c1 = *(const float4*)(qp + d0 + 4);
            float4 c2 = *(const float4*)(qp + d0 + 16);
            float4 c3 = *(const float4*)(qp + d0 + 20);
            float cv[16];
#pragma unroll
            for (int j = 0; j < 4; ++j) {
                cv[j]      = ((const float*)&c0)[j];
                cv[4 + j]  = ((const float*)&c1)[j];
                cv[8 + j]  = ((const float*)&c2)[j];
                cv[12 + j] = ((const float*)&c3)[j];
            }
            Frag16 f;
#pragma unroll
            for (int j = 0; j < 8; ++j)
                f.w[j] = f2bf2(cv[2 * j], cv[2 * j + 1]);
            aq[s] = f.v;
        }
    }

    v8f o[4];
#pragma unroll
    for (int dt = 0; dt < 4; ++dt)
        o[dt] = v8f{0.f, 0.f, 0.f, 0.f, 0.f, 0.f, 0.f, 0.f};
    float m_run[8], l_run[8];
#pragma unroll
    for (int i = 0; i < 8; ++i) { m_run[i] = -1e30f; l_run[i] = 0.f; }

    const int pbase = w * 16 * 64;

    // staging mapping: thread owns 2 adjacent k' rows x 16 d values
    const int kr2 = (tid >> 2) * 2;                 // 0..62
    const int dg  = (tid & 3) * 16;                 // 0/16/32/48
    unsigned int* K32 = (unsigned int*)Kls;
    unsigned int* V32 = (unsigned int*)Vls;

    for (int kt = 0; kt < 2048; kt += 64) {
        // ---- stage K (row-major) and V (col-major) tiles as bf16 ----
        const float* kp = Kb + base_bh + (size_t)(kt + kr2) * 1024 + dg;
        const float* vp = Vb + base_bh + (size_t)(kt + kr2) * 1024 + dg;
        float4 k0[4], k1[4], v0[4], v1[4];
#pragma unroll
        for (int j2 = 0; j2 < 4; ++j2) {
            k0[j2] = *(const float4*)(kp + j2 * 4);
            k1[j2] = *(const float4*)(kp + 1024 + j2 * 4);
            v0[j2] = *(const float4*)(vp + j2 * 4);
            v1[j2] = *(const float4*)(vp + 1024 + j2 * 4);
        }
#pragma unroll
        for (int j = 0; j < 8; ++j) {
            K32[(kr2 * 64 + dg) / 2 + j] =
                f2bf2(((const float*)k0)[2 * j], ((const float*)k0)[2 * j + 1]);
            K32[((kr2 + 1) * 64 + dg) / 2 + j] =
                f2bf2(((const float*)k1)[2 * j], ((const float*)k1)[2 * j + 1]);
        }
#pragma unroll
        for (int j = 0; j < 16; ++j)
            V32[((dg + j) * 64 + kr2) / 2] =
                f2bf2(((const float*)v0)[j], ((const float*)v1)[j]);
        __syncthreads();

        // ---- scores S = Q @ K^T  (contraction over d = 2 WMMA k-steps) ----
        v8f sc[4];
#pragma unroll
        for (int nt = 0; nt < 4; ++nt) {
            int col = nt * 16 + nlan;               // k' column in this tile
            v8f c = v8f{0.f, 0.f, 0.f, 0.f, 0.f, 0.f, 0.f, 0.f};
#pragma unroll
            for (int s = 0; s < 2; ++s) {
                int kb = s * 32 + half8;
                Frag16 f;
                f.q[0] = *(const uint4*)&Kls[col * 64 + kb];
                f.q[1] = *(const uint4*)&Kls[col * 64 + kb + 16];
                c = __builtin_amdgcn_wmma_f32_16x16x32_bf16(
                        false, aq[s], false, f.v, (short)0, c, false, false);
            }
            sc[nt] = c;
        }

        // ---- online softmax (rows live in 16-lane halves -> xor 1,2,4,8) ----
#pragma unroll
        for (int i = 0; i < 8; ++i) {
            float rm = fmaxf(fmaxf(sc[0][i], sc[1][i]), fmaxf(sc[2][i], sc[3][i]));
            rm = fmaxf(rm, __shfl_xor(rm, 1));
            rm = fmaxf(rm, __shfl_xor(rm, 2));
            rm = fmaxf(rm, __shfl_xor(rm, 4));
            rm = fmaxf(rm, __shfl_xor(rm, 8));
            float mn    = fmaxf(m_run[i], rm);
            float scale = __expf(m_run[i] - mn);
            float rs = 0.f;
#pragma unroll
            for (int nt = 0; nt < 4; ++nt) {
                float p = __expf(sc[nt][i] - mn);
                sc[nt][i] = p;
                rs += p;
            }
            rs += __shfl_xor(rs, 1);
            rs += __shfl_xor(rs, 2);
            rs += __shfl_xor(rs, 4);
            rs += __shfl_xor(rs, 8);
            l_run[i] = l_run[i] * scale + rs;
            m_run[i] = mn;
#pragma unroll
            for (int dt = 0; dt < 4; ++dt)
                o[dt][i] = o[dt][i] * scale;
        }

        // ---- C-layout -> A-layout for P via wave-private LDS slab ----
#pragma unroll
        for (int nt = 0; nt < 4; ++nt)
#pragma unroll
            for (int i = 0; i < 8; ++i)
                Pls[pbase + (i + half8) * 64 + nt * 16 + nlan] = f2bf(sc[nt][i]);

        // ---- O += P @ V  (contraction over k' = 2 WMMA k-steps) ----
        v16bf ap[2];
#pragma unroll
        for (int s = 0; s < 2; ++s) {
            int ka = s * 32 + half8;
            Frag16 f;
            f.q[0] = *(const uint4*)&Pls[pbase + nlan * 64 + ka];
            f.q[1] = *(const uint4*)&Pls[pbase + nlan * 64 + ka + 16];
            ap[s] = f.v;
        }
#pragma unroll
        for (int dt = 0; dt < 4; ++dt) {
#pragma unroll
            for (int s = 0; s < 2; ++s) {
                int kb = s * 32 + half8;
                Frag16 f;
                f.q[0] = *(const uint4*)&Vls[(dt * 16 + nlan) * 64 + kb];
                f.q[1] = *(const uint4*)&Vls[(dt * 16 + nlan) * 64 + kb + 16];
                o[dt] = __builtin_amdgcn_wmma_f32_16x16x32_bf16(
                            false, ap[s], false, f.v, (short)0, o[dt], false, false);
            }
        }
        __syncthreads();                            // before next tile overwrite
    }

    // ---- normalize and store ----
#pragma unroll
    for (int dt = 0; dt < 4; ++dt) {
#pragma unroll
        for (int i = 0; i < 8; ++i) {
            int m = i + half8;
            Ob[base_bh + (size_t)(q0 + m) * 1024 + dt * 16 + nlan] =
                o[dt][i] / l_run[i];
        }
    }
}

// ---------------------------------------------------------------------------
extern "C" void kernel_launch(void* const* d_in, const int* in_sizes, int n_in,
                              void* d_out, int out_size, void* d_ws, size_t ws_size,
                              hipStream_t stream) {
    (void)in_sizes; (void)n_in; (void)out_size; (void)ws_size;
    const float* q  = (const float*)d_in[0];
    const float* k  = (const float*)d_in[1];
    const float* v  = (const float*)d_in[2];
    const float* Wq = (const float*)d_in[3];
    const float* bq = (const float*)d_in[4];
    const float* Wk = (const float*)d_in[5];
    const float* bk = (const float*)d_in[6];
    const float* Wv = (const float*)d_in[7];
    const float* bv = (const float*)d_in[8];
    const float* Wo = (const float*)d_in[9];
    const float* bo = (const float*)d_in[10];
    float* out = (float*)d_out;

    const int M = 2 * 2048;     // B * S
    const int D = 1024;

    float* wsf = (float*)d_ws;
    float* Qp = wsf;
    float* Kp = wsf + (size_t)M * D;
    float* Vp = wsf + 2 * (size_t)M * D;
    float* Ap = wsf + 3 * (size_t)M * D;

    dim3 ggrid(D / TN, M / TM);
    // fold 1/sqrt(head_dim)=0.125 into the Q projection
    gemm_bias_bf16<<<ggrid, 256, 0, stream>>>(q, Wq, bq, Qp, M, D, D, 0.125f);
    gemm_bias_bf16<<<ggrid, 256, 0, stream>>>(k, Wk, bk, Kp, M, D, D, 1.0f);
    gemm_bias_bf16<<<ggrid, 256, 0, stream>>>(v, Wv, bv, Vp, M, D, D, 1.0f);

    attn_kernel<<<dim3(2048 / 64, 16, 2), 128, 0, stream>>>(Qp, Kp, Vp, Ap);

    gemm_bias_bf16<<<ggrid, 256, 0, stream>>>(Ap, Wo, bo, out, M, D, D, 1.0f);
}
